// NT_Xent_6528350290390
// MI455X (gfx1250) — matrix operand order repaired and checked by reading, not compiled, not run on hardware
//
#include <hip/hip_runtime.h>
#include <hip/hip_bf16.h>

typedef __attribute__((ext_vector_type(16))) __bf16 v16bf;
typedef __attribute__((ext_vector_type(8)))  __bf16 v8bf;
typedef __attribute__((ext_vector_type(8)))  float  v8f;
typedef __attribute__((ext_vector_type(4)))  int    v4i;

#define DIM 512
#define LDS_STRIDE 520      // 512 + 8 bf16 pad -> conflict-free column access
#define ROWS_PER_WAVE 32    // two 16-row A tiles in registers
#define WAVES 8
#define ROWS_PER_BLOCK (ROWS_PER_WAVE * WAVES)   // 256
#define NSPLIT 8            // column-range splits (grid.y)

#if __has_builtin(__builtin_amdgcn_global_load_async_to_lds_b128)
#define HAVE_ASYNC_LDS 1
#else
#define HAVE_ASYNC_LDS 0
#endif

__device__ __forceinline__ unsigned short f2bf_rne(float f) {
    unsigned int u = __float_as_uint(f);
    u += 0x7fffu + ((u >> 16) & 1u);
    return (unsigned short)(u >> 16);
}
__device__ __forceinline__ float bf2f(unsigned short h) {
    return __uint_as_float(((unsigned int)h) << 16);
}

// ---------------------------------------------------------------------------
// Kernel 1: per-row L2 normalize (fp32 math, reference eps) -> bf16
// ---------------------------------------------------------------------------
__global__ void nt_xent_normalize(const float* __restrict__ left,
                                  const float* __restrict__ right,
                                  unsigned short* __restrict__ zl,
                                  unsigned short* __restrict__ zr,
                                  int N) {
    __shared__ float sdata[256];
    int row = blockIdx.x;
    const float* src;
    unsigned short* dst;
    if (row < N) { src = left  + (size_t)row * DIM;       dst = zl + (size_t)row * DIM; }
    else         { src = right + (size_t)(row - N) * DIM; dst = zr + (size_t)(row - N) * DIM; }
    int t = threadIdx.x;
    float x0 = src[t], x1 = src[t + 256];
    sdata[t] = x0 * x0 + x1 * x1;
    __syncthreads();
    for (int s = 128; s > 0; s >>= 1) {
        if (t < s) sdata[t] += sdata[t + s];
        __syncthreads();
    }
    float inv = 1.0f / fmaxf(sqrtf(sdata[0]), 1e-12f);
    dst[t]       = f2bf_rne(x0 * inv);
    dst[t + 256] = f2bf_rne(x1 * inv);
}

// ---------------------------------------------------------------------------
// Kernel 2: fused GEMM + exp row-sum partials.
// grid = (N/256, NSPLIT). Block = 8 waves; wave owns 32 rows (2 A tiles in
// 256 VGPRs). B tiles (16 cols x 512 bf16) TRIPLE-buffered in LDS, staged
// two tiles ahead with GLOBAL_LOAD_ASYNC_TO_LDS_B128 (ASYNCcnt-tracked,
// in-order retire -> s_wait_asynccnt 4 drains exactly the oldest tile).
// Per 16-col tile: 2x16 v_wmma_f32_16x16x32_bf16 with B fragments rotated
// through 3 register buffers (DS prefetch depth 2). Deterministic per-split
// partial row sums of exp(2*s) over the combined [Zl | Zr] column range.
// ---------------------------------------------------------------------------
__global__ void __launch_bounds__(256, 1)
nt_xent_main(const unsigned short* __restrict__ zl,
             const unsigned short* __restrict__ zr,
             float* __restrict__ part,        // [NSPLIT][N]
             int N) {
    __shared__ __align__(16) unsigned short sB[3][16 * LDS_STRIDE];

    const int tid   = threadIdx.x;
    const int w     = tid >> 5;
    const int l     = tid & 31;
    const int m16   = l & 15;
    const int khalf = l >> 4;
    const int split = blockIdx.y;
    const int row0  = blockIdx.x * ROWS_PER_BLOCK + w * ROWS_PER_WAVE;

    const int tps    = (N >> 4) / NSPLIT;   // tiles per split per pass (64)
    const int jt0    = split * tps;
    const int ntiles = 2 * tps;             // both passes, pipelined together

    union Frag { v16bf v; v8bf h[2]; };

    // staging-chunk coordinates for this thread (4 x 16B per tile)
    int srow[4], scol[4];
    #pragma unroll
    for (int i = 0; i < 4; ++i) {
        int ch = tid + 256 * i;
        srow[i] = ch >> 6;
        scol[i] = (ch & 63) * 8;
    }

    auto tileSrc = [&](int t) -> const unsigned short* {
        int p = (t >= tps) ? 1 : 0;
        int j = jt0 + (t - (p ? tps : 0));
        return (p ? zr : zl) + (size_t)j * 16 * DIM;
    };

    // stage one 16x512 bf16 tile into LDS buffer `buf`
    auto stage = [&](const unsigned short* srcTile, int buf) {
#if HAVE_ASYNC_LDS
        #pragma unroll
        for (int i = 0; i < 4; ++i) {
            __builtin_amdgcn_global_load_async_to_lds_b128(
                (__attribute__((address_space(1))) v4i*)(srcTile + (size_t)(tid + 256 * i) * 8),
                (__attribute__((address_space(3))) v4i*)(&sB[buf][srow[i] * LDS_STRIDE + scol[i]]),
                0, 0);
        }
#else
        const uint4* gsrc = (const uint4*)srcTile;
        uint4 g[4];
        #pragma unroll
        for (int i = 0; i < 4; ++i) g[i] = gsrc[tid + 256 * i];
        #pragma unroll
        for (int i = 0; i < 4; ++i)
            *(uint4*)(&sB[buf][srow[i] * LDS_STRIDE + scol[i]]) = g[i];
#endif
    };

    // ---- two 16x512 A tiles into registers (exact WMMA A layout) ----
    v16bf aA[2][16];
    #pragma unroll
    for (int u = 0; u < 2; ++u) {
        const unsigned short* ap =
            zl + (size_t)(row0 + u * 16 + m16) * DIM + khalf * 8;
        #pragma unroll
        for (int s = 0; s < 16; ++s) {
            Frag f;
            f.h[0] = *(const v8bf*)(ap + s * 32);
            f.h[1] = *(const v8bf*)(ap + s * 32 + 16);
            aA[u][s] = f.v;
        }
    }

    float acc[2][8];   // [u][r] : refl+between merged row sums
    #pragma unroll
    for (int u = 0; u < 2; ++u)
        #pragma unroll
        for (int r = 0; r < 8; ++r) acc[u][r] = 0.0f;

    // prologue: stage tiles 0 and 1 (prefetch depth 2)
    stage(tileSrc(0), 0);
    stage(tileSrc(1), 1);

    int bcur = 0, bmid = 1, bnxt = 2;
    for (int tt = 0; tt < ntiles; ++tt) {
#if HAVE_ASYNC_LDS
        // async loads retire in order: <=4 outstanding leaves only tile tt+1's
        // chunks in flight, i.e. tile tt has fully landed in LDS.
        if (tt + 1 < ntiles) __builtin_amdgcn_s_wait_asynccnt(4);
        else                 __builtin_amdgcn_s_wait_asynccnt(0);
#endif
        __syncthreads();   // all waves: buf[bcur] staged; reads of buf[bnxt] done

        // background-stage tile tt+2 two iterations ahead
        if (tt + 2 < ntiles) stage(tileSrc(tt + 2), bnxt);

        // ---- compute: 16 K-slabs, B frags rotated through 3 reg buffers ----
        v8f c0 = {}, c1 = {};
        const unsigned short* bp = &sB[bcur][m16 * LDS_STRIDE + khalf * 8];
        Frag fb[3];
        #pragma unroll
        for (int s = 0; s < 2; ++s) {
            fb[s].h[0] = *(const v8bf*)(bp + s * 32);
            fb[s].h[1] = *(const v8bf*)(bp + s * 32 + 16);
        }
        #pragma unroll
        for (int s = 0; s < 16; ++s) {
            if (s + 2 < 16) {
                fb[(s + 2) % 3].h[0] = *(const v8bf*)(bp + (s + 2) * 32);
                fb[(s + 2) % 3].h[1] = *(const v8bf*)(bp + (s + 2) * 32 + 16);
            }
            c0 = __builtin_amdgcn_wmma_f32_16x16x32_bf16(
                     false, aA[0][s], false, fb[s % 3].v, (short)0, c0, false, false);
            c1 = __builtin_amdgcn_wmma_f32_16x16x32_bf16(
                     false, aA[1][s], false, fb[s % 3].v, (short)0, c1, false, false);
        }

        // fused epilogue: acc += exp(s/tau), tau = 0.5
        #pragma unroll
        for (int r = 0; r < 8; ++r) {
            acc[0][r] += __expf(c0[r] * 2.0f);
            acc[1][r] += __expf(c1[r] * 2.0f);
        }

        int tmp = bcur; bcur = bmid; bmid = bnxt; bnxt = tmp;
    }

    // reduce row sums across each 16-lane half (columns of the C tile)
    #pragma unroll
    for (int u = 0; u < 2; ++u)
        #pragma unroll
        for (int r = 0; r < 8; ++r)
            #pragma unroll
            for (int m = 1; m < 16; m <<= 1)
                acc[u][r] += __shfl_xor(acc[u][r], m, 32);

    // lanes 0..7 -> rows 0..7 of each tile; lanes 16..23 -> rows 8..15
    int mlocal = -1;
    if (l < 8) mlocal = l;
    else if (l >= 16 && l < 24) mlocal = l - 8;
    if (mlocal >= 0) {
        int rsel = mlocal & 7;
        float s0 = 0.f, s1 = 0.f;
        #pragma unroll
        for (int r = 0; r < 8; ++r)
            if (rsel == r) { s0 = acc[0][r]; s1 = acc[1][r]; }
        size_t basep = (size_t)split * N;
        part[basep + row0 + mlocal]      = s0;
        part[basep + row0 + 16 + mlocal] = s1;
    }
}

// ---------------------------------------------------------------------------
// Kernel 3: finalize. One wave per row: diagonals via bf16 dot, sum partials,
// loss = log(sum_refl + sum_btw - exp(2*d_ll)) - 2*d_lr
// ---------------------------------------------------------------------------
__global__ void nt_xent_finalize(const unsigned short* __restrict__ zl,
                                 const unsigned short* __restrict__ zr,
                                 const float* __restrict__ part,
                                 float* __restrict__ out, int N) {
    int row = blockIdx.x * 8 + (threadIdx.x >> 5);
    int l   = threadIdx.x & 31;
    const unsigned short* pl = zl + (size_t)row * DIM + l * 16;
    const unsigned short* pr = zr + (size_t)row * DIM + l * 16;
    float dll = 0.f, dlr = 0.f;
    #pragma unroll
    for (int k = 0; k < 16; ++k) {
        float a = bf2f(pl[k]);
        float b = bf2f(pr[k]);
        dll += a * a;
        dlr += a * b;
    }
    #pragma unroll
    for (int m = 1; m < 32; m <<= 1) {
        dll += __shfl_xor(dll, m, 32);
        dlr += __shfl_xor(dlr, m, 32);
    }
    if (l == 0) {
        float s = 0.f;
        #pragma unroll
        for (int sp = 0; sp < NSPLIT; ++sp)
            s += part[(size_t)sp * N + row];
        float denom = s - __expf(2.0f * dll);
        out[row] = __logf(denom) - 2.0f * dlr;
    }
}

// ---------------------------------------------------------------------------
extern "C" void kernel_launch(void* const* d_in, const int* in_sizes, int n_in,
                              void* d_out, int out_size, void* d_ws, size_t ws_size,
                              hipStream_t stream) {
    const float* left  = (const float*)d_in[0];
    const float* right = (const float*)d_in[1];
    int N = in_sizes[0] / DIM;                               // 8192

    unsigned short* zl = (unsigned short*)d_ws;              // N*512 bf16
    unsigned short* zr = zl + (size_t)N * DIM;
    float* part = (float*)(zr + (size_t)N * DIM);            // [NSPLIT][N]
    float* out = (float*)d_out;

    nt_xent_normalize<<<2 * N, 256, 0, stream>>>(left, right, zl, zr, N);
    dim3 grid(N / ROWS_PER_BLOCK, NSPLIT);
    nt_xent_main<<<grid, 256, 0, stream>>>(zl, zr, part, N);
    nt_xent_finalize<<<N / 8, 256, 0, stream>>>(zl, zr, part, out, N);
}
